// FTLayer_88433376625443
// MI455X (gfx1250) — compile-verified, matrix operand order checked by reading, not run on hardware
//
#include <hip/hip_runtime.h>
#include <hip/hip_bf16.h>

// ---------------- CDNA5 WMMA types ----------------
typedef __attribute__((ext_vector_type(16))) __bf16 v16bf;
typedef __attribute__((ext_vector_type(8)))  float  v8f;

union FragBF { v16bf v; unsigned int u[8]; };

__device__ __forceinline__ v8f wmma_bf16(v16bf a, v16bf b, v8f c) {
    // 8 args: (neg_a, A, neg_b, B, c_mod, C, reuse_a, reuse_b)
    return __builtin_amdgcn_wmma_f32_16x16x32_bf16(false, a, false, b, (short)0, c, false, false);
}

// shifted softplus: softplus(v) - log(2)
__device__ __forceinline__ float sspf(float v) {
    float a  = __builtin_fabsf(v);
    float sp = fmaxf(v, 0.0f) + __logf(1.0f + __expf(-a));
    return sp - 0.6931471824645996f;
}

__device__ __forceinline__ unsigned short f2bf(float x) {
    union { float f; unsigned int u; } c; c.f = x;
    unsigned int r = c.u + 0x7FFFu + ((c.u >> 16) & 1u);   // round-to-nearest-even
    return (unsigned short)(r >> 16);
}
__device__ __forceinline__ unsigned int pack2(float a, float b) {
    return (unsigned int)f2bf(a) | ((unsigned int)f2bf(b) << 16);
}

// 16x32 bf16 fragment from row-major source (stride 256 bf16).
// lane&15 = row, lane>>4 = K-half per ISA 16-bit A/B VGPR layout.
__device__ __forceinline__ v16bf load_frag_row(const unsigned short* __restrict__ base,
                                               int m, int half, int k0) {
    FragBF f;
#pragma unroll
    for (int v = 0; v < 8; ++v) {
        int k = k0 + ((v < 4) ? 0 : 16) + half * 8 + (v & 3) * 2;
        f.u[v] = *(const unsigned int*)(base + m * 256 + k);
    }
    return f.v;
}
__device__ __forceinline__ v16bf load_frag_lds(const unsigned short* base,
                                               int m, int half, int k0) {
    FragBF f;
#pragma unroll
    for (int v = 0; v < 8; ++v) {
        int k = k0 + ((v < 4) ? 0 : 16) + half * 8 + (v & 3) * 2;
        f.u[v] = *(const unsigned int*)(base + m * 256 + k);
    }
    return f.v;
}

// ---------------- constants ----------------
#define A_DIM 512
#define N_DIM 128
#define FDIM  256

// ---------------- K0: convert+transpose 3 weights to bf16 [n][k] ----------------
__global__ void __launch_bounds__(256)
ft_prep(const float* __restrict__ in2f_w, const float* __restrict__ fw2,
        const float* __restrict__ f2out_w,
        unsigned short* __restrict__ wInT, unsigned short* __restrict__ wF2T,
        unsigned short* __restrict__ wOutT) {
    int id    = blockIdx.x * 256 + threadIdx.x;       // 0 .. 3*65536
    int which = id >> 16;
    int rem   = id & 65535;
    int n     = rem >> 8;
    int k     = rem & 255;
    const float* src = (which == 0) ? in2f_w : (which == 1) ? fw2 : f2out_w;
    unsigned short* dst = (which == 0) ? wInT : (which == 1) ? wF2T : wOutT;
    dst[n * 256 + k] = f2bf(src[k * 256 + n]);        // transpose: wT[n][k] = w[k][n]
}

// ---------------- K1: facts = x @ in2f_w + b  (WMMA, f32 out to ws) ----------------
__global__ void __launch_bounds__(128)
ft_facts(const float* __restrict__ x, const float* __restrict__ in2f_b,
         const unsigned short* __restrict__ wInT, float* __restrict__ facts) {
    __shared__ unsigned short Xbuf[4 * 16 * FDIM];    // 32 KB
    const int lane = threadIdx.x & 31;
    const int wave = threadIdx.x >> 5;
    const int half = lane >> 4;
    const int ln   = lane & 15;
    const int m0   = (blockIdx.x * 4 + wave) * 16;    // row block in [0, 2048)
    unsigned short* X = Xbuf + wave * 16 * FDIM;

#pragma unroll 4
    for (int row = 0; row < 16; ++row) {
        const float* src = x + (m0 + row) * FDIM + lane * 8;
        float h[8];
#pragma unroll
        for (int j = 0; j < 8; ++j) h[j] = src[j];
        uint4 p;
        p.x = pack2(h[0], h[1]); p.y = pack2(h[2], h[3]);
        p.z = pack2(h[4], h[5]); p.w = pack2(h[6], h[7]);
        *(uint4*)(X + row * FDIM + lane * 8) = p;     // one ds_store_b128
    }

    for (int n0 = 0; n0 < FDIM; n0 += 16) {
        v8f acc = {};
#pragma unroll
        for (int k0 = 0; k0 < FDIM; k0 += 32) {
            v16bf a = load_frag_lds(X, ln, half, k0);
            v16bf b = load_frag_row(wInT, n0 + ln, half, k0);
            acc = wmma_bf16(a, b, acc);
        }
        int   f    = n0 + ln;
        float bias = in2f_b[f];
#pragma unroll
        for (int j = 0; j < 8; ++j)
            facts[(m0 + j + half * 8) * FDIM + f] = acc[j] + bias;
    }
}

// ---- K2: fused, 32-row strips; block-shared weight tiles staged in LDS --------
__global__ void __launch_bounds__(128)
ft_main(const float* __restrict__ r,            // (B,A,N)
        const int* __restrict__ nbr,            // (B,A,N)
        const float* __restrict__ fw1, const float* __restrict__ fb1,
        const float* __restrict__ fb2, const float* __restrict__ f2out_b,
        const float* __restrict__ facts,        // ws f32 (B*A, 256)
        const unsigned short* __restrict__ wF2T,
        const unsigned short* __restrict__ wOutT,
        float* __restrict__ out) {
    __shared__ unsigned short Hbuf[4 * 32 * FDIM];   // 64 KB (per-wave strips)
    __shared__ unsigned short Ybuf[4 * 32 * FDIM];   // 64 KB (per-wave strips)
    __shared__ unsigned short Wbuf[16 * FDIM];       //  8 KB (block-shared B tile)
    const int lane  = threadIdx.x & 31;
    const int wave  = threadIdx.x >> 5;
    const int half  = lane >> 4;
    const int ln    = lane & 15;
    const int strip = blockIdx.x * 4 + wave;
    const int m0    = strip * 32;                    // 32 consecutive n of one (b,a)
    const int pa    = m0 >> 7;                       // b*A + a   (N=128)
    const int b     = pa >> 9;                       // / A (A=512)
    unsigned short* H = Hbuf + wave * 32 * FDIM;
    unsigned short* Y = Ybuf + wave * 32 * FDIM;

    // ---- Stage 1: H[row][col] = ssp(r[row]*fw1[col] + fb1[col]), bf16 in LDS ----
    float w1c[8], b1c[8];
#pragma unroll
    for (int j = 0; j < 8; ++j) {
        w1c[j] = fw1[lane * 8 + j];
        b1c[j] = fb1[lane * 8 + j];
    }
    for (int row = 0; row < 32; ++row) {
        float rv = r[m0 + row];
        float h[8];
#pragma unroll
        for (int j = 0; j < 8; ++j) h[j] = sspf(rv * w1c[j] + b1c[j]);
        uint4 p;
        p.x = pack2(h[0], h[1]); p.y = pack2(h[2], h[3]);
        p.z = pack2(h[4], h[5]); p.w = pack2(h[6], h[7]);
        *(uint4*)(H + row * FDIM + lane * 8) = p;    // one ds_store_b128
    }

    // per-strip gather bases for the two 16-row tiles (8 rows per lane-half each)
    int nb0[8], nb1[8];
#pragma unroll
    for (int j = 0; j < 8; ++j) {
        nb0[j] = (b * A_DIM + nbr[m0 + j + half * 8]) * FDIM;
        nb1[j] = (b * A_DIM + nbr[m0 + 16 + j + half * 8]) * FDIM;
    }
    const float* xiRow = facts + pa * FDIM;

    // ---- Stage 2: W = H @ fw2 + fb2 ; y = xi*W*xj -> Ybuf (bf16) ----
    for (int n0 = 0; n0 < FDIM; n0 += 16) {
        __syncthreads();                              // previous tile fully consumed
        {   // cooperative stage of 16x256 bf16 weight tile (8 KB): 4 x uint4/thread
            const uint4* src = (const uint4*)(wF2T + n0 * FDIM);
            uint4*       dst = (uint4*)Wbuf;
#pragma unroll
            for (int t = 0; t < 4; ++t)
                dst[threadIdx.x + t * 128] = src[threadIdx.x + t * 128];
        }
        __syncthreads();
        v8f acc0 = {}, acc1 = {};
#pragma unroll
        for (int k0 = 0; k0 < FDIM; k0 += 32) {
            v16bf bf = load_frag_lds(Wbuf, ln, half, k0);        // shared by block,
            v16bf a0 = load_frag_lds(H, ln, half, k0);           // used twice per wave
            v16bf a1 = load_frag_lds(H + 16 * FDIM, ln, half, k0);
            acc0 = wmma_bf16(a0, bf, acc0);
            acc1 = wmma_bf16(a1, bf, acc1);
        }
        int   f    = n0 + ln;
        float bias = fb2[f];
        float xi   = xiRow[f];
#pragma unroll
        for (int j = 0; j < 8; ++j)
            Y[(j + half * 8) * FDIM + f] =
                f2bf(xi * (acc0[j] + bias) * facts[nb0[j] + f]);
#pragma unroll
        for (int j = 0; j < 8; ++j)
            Y[(16 + j + half * 8) * FDIM + f] =
                f2bf(xi * (acc1[j] + bias) * facts[nb1[j] + f]);
    }

    // ---- Stage 3: out = ssp(y @ f2out_w + b) ----
    for (int n0 = 0; n0 < FDIM; n0 += 16) {
        __syncthreads();
        {
            const uint4* src = (const uint4*)(wOutT + n0 * FDIM);
            uint4*       dst = (uint4*)Wbuf;
#pragma unroll
            for (int t = 0; t < 4; ++t)
                dst[threadIdx.x + t * 128] = src[threadIdx.x + t * 128];
        }
        __syncthreads();
        v8f acc0 = {}, acc1 = {};
#pragma unroll
        for (int k0 = 0; k0 < FDIM; k0 += 32) {
            v16bf bf = load_frag_lds(Wbuf, ln, half, k0);
            v16bf a0 = load_frag_lds(Y, ln, half, k0);
            v16bf a1 = load_frag_lds(Y + 16 * FDIM, ln, half, k0);
            acc0 = wmma_bf16(a0, bf, acc0);
            acc1 = wmma_bf16(a1, bf, acc1);
        }
        int   f    = n0 + ln;
        float bias = f2out_b[f];
#pragma unroll
        for (int j = 0; j < 8; ++j)
            __builtin_nontemporal_store(sspf(acc0[j] + bias),
                                        out + (m0 + j + half * 8) * FDIM + f);
#pragma unroll
        for (int j = 0; j < 8; ++j)
            __builtin_nontemporal_store(sspf(acc1[j] + bias),
                                        out + (m0 + 16 + j + half * 8) * FDIM + f);
    }
}

// ---------------- launch ----------------
extern "C" void kernel_launch(void* const* d_in, const int* in_sizes, int n_in,
                              void* d_out, int out_size, void* d_ws, size_t ws_size,
                              hipStream_t stream) {
    const float* x       = (const float*)d_in[0];
    const float* r_ij    = (const float*)d_in[1];
    // d_in[2] = pairwise_mask (unused by reference math)
    const float* in2f_w  = (const float*)d_in[3];
    const float* in2f_b  = (const float*)d_in[4];
    const float* f2out_w = (const float*)d_in[5];
    const float* f2out_b = (const float*)d_in[6];
    const float* fw1     = (const float*)d_in[7];
    const float* fb1     = (const float*)d_in[8];
    const float* fw2     = (const float*)d_in[9];
    const float* fb2     = (const float*)d_in[10];
    const int*   nbr     = (const int*)d_in[11];

    char* ws = (char*)d_ws;
    float*          factsWS = (float*)(ws);                         // 2048*256*4 = 2 MB
    unsigned short* wInT    = (unsigned short*)(ws + 2097152);      // 128 KB bf16 [n][k]
    unsigned short* wF2T    = (unsigned short*)(ws + 2097152 + 131072);
    unsigned short* wOutT   = (unsigned short*)(ws + 2097152 + 262144);

    // K0: weight convert+transpose (3 * 65536 elements)
    ft_prep<<<768, 256, 0, stream>>>(in2f_w, fw2, f2out_w, wInT, wF2T, wOutT);

    // K1: facts = x @ in2f_w + b   (2048 rows -> 128 strips, 4 strips/block)
    ft_facts<<<32, 128, 0, stream>>>(x, in2f_b, wInT, factsWS);

    // K2: fused pairwise layer (262144 rows -> 8192 strips of 32, 4 strips/block)
    ft_main<<<2048, 128, 0, stream>>>(r_ij, nbr, fw1, fb1, fb2, f2out_b,
                                      factsWS, wF2T, wOutT, (float*)d_out);
}
